// MambaSequenceClassifierMemoryCNN_88459146428786
// MI455X (gfx1250) — matrix operand order repaired and checked by reading, not compiled, not run on hardware
//
// MI455X (gfx1250) fused Mamba sequence classifier forward pass.
// Round 2: big GEMMs on v_wmma_f32_16x16x32_bf16 (confirmed in histogram);
// this round adds Tensor Data Mover (tensor_load_to_lds) staging of the
// bf16 weight tiles into LDS, with s_wait_tensorcnt synchronization.
//
// Workspace requirement: ~97 MB (see layout in kernel_launch).

#include <hip/hip_runtime.h>

#define BATCH 8
#define L_IN  8192
#define L0    4096
#define LH    2048
#define DM    128
#define DI    256
#define DS    16
#define DTR   8
#define NL    4
#define NC    100
#define ROWS  (BATCH * LH)   // 16384

typedef __attribute__((ext_vector_type(16))) __bf16 v16bf;
typedef __attribute__((ext_vector_type(8)))  float  v8f;
typedef unsigned int u32x4 __attribute__((ext_vector_type(4)));
typedef unsigned int u32x8 __attribute__((ext_vector_type(8)));

// ---------------- helpers ----------------
__device__ __forceinline__ unsigned short bf16bits(float f) {
  unsigned u = __builtin_bit_cast(unsigned, f);
  u += 0x7fffu + ((u >> 16) & 1u);          // round-to-nearest-even
  return (unsigned short)(u >> 16);
}
__device__ __forceinline__ float sigmoidf_(float x) { return 1.0f / (1.0f + __expf(-x)); }
__device__ __forceinline__ float siluf_(float x)    { return x * sigmoidf_(x); }
__device__ __forceinline__ float softplusf_(float x){ return (x > 20.0f) ? x : log1pf(__expf(x)); }

// ---------------- TDM: DMA a strided 2-D bf16 tile from global into LDS ----
// D# layout per CDNA5 ISA 8.3/8.4. Elements are 2 bytes (data_size=1).
// tile_d0 = elements per row, tile_d1 = rows, stride_elems = global row stride.
// LDS destination gets pad_amount dwords of padding after each pad_interval
// dwords (codes per ISA: interval 5 -> 64 DW = 256B row; amount 3 -> 4 DW = 16B),
// matching the [rows][d0+8] padded __shared__ tiles used below.
__device__ __forceinline__ void tdm_load_tile_bf16(
    const void* gsrc, const void* lds_dst,
    unsigned tile_d0, unsigned tile_d1, unsigned stride_elems,
    unsigned pad_interval_code, unsigned pad_amount_code, unsigned pad_en) {
  unsigned long long ga = (unsigned long long)gsrc;
  unsigned la = (unsigned)(unsigned long long)lds_dst;  // LDS aperture: low 32b = LDS byte offset
  u32x4 g0;
  g0[0] = 1u;                                              // count=1 valid user descriptor
  g0[1] = la;                                              // lds_addr (bytes)
  g0[2] = (unsigned)ga;                                    // global_addr[31:0]
  g0[3] = (unsigned)((ga >> 32) & 0x01ffffffu) | (2u << 30); // global_addr[56:32] | type=2
  unsigned td0 = stride_elems;          // tensor_dim0 = full row length
  unsigned td1 = 1u << 20;              // tensor_dim1 large: no OOB clipping of tile rows
  unsigned long long s0 = stride_elems; // tensor_dim0_stride (data_size units)
  u32x8 g1;
  g1[0] = (1u << 16)                    // data_size = 1 (2 bytes)
        | (pad_en << 20) | (pad_interval_code << 22) | (pad_amount_code << 25);
  g1[1] = (td0 & 0xffffu) << 16;                       // tensor_dim0[15:0]
  g1[2] = (td0 >> 16) | ((td1 & 0xffffu) << 16);       // tensor_dim0[31:16] | tensor_dim1[15:0]
  g1[3] = (td1 >> 16) | ((tile_d0 & 0xffffu) << 16);   // tensor_dim1[31:16] | tile_dim0
  g1[4] = (tile_d1 & 0xffffu);                         // tile_dim1 | tile_dim2=0
  g1[5] = (unsigned)s0;                                // tensor_dim0_stride[31:0]
  g1[6] = (unsigned)((s0 >> 32) & 0xffffu);            // stride[47:32] | dim1_stride[15:0]=0
  g1[7] = 0u;
  // 2-D tensor: VADDR2/VADDR3 omitted (NULL). Tracked by TENSORcnt.
  asm volatile("tensor_load_to_lds %0, %1" :: "s"(g0), "s"(g1) : "memory");
}

// ---------------- weight convert f32 -> bf16 ----------------
__global__ void cvt_bf16(const float* __restrict__ src, unsigned short* __restrict__ dst, int n) {
  int i = blockIdx.x * blockDim.x + threadIdx.x;
  if (i < n) dst[i] = bf16bits(src[i]);
}

__global__ void mem_init_k(const float* __restrict__ mi, float* __restrict__ mem) {
  int i = blockIdx.x * blockDim.x + threadIdx.x;
  if (i < BATCH * DM) mem[i] = mi[i & (DM - 1)];
}

// ---------------- stem convs ----------------
__global__ void stem_conv0(const float* __restrict__ x, const float* __restrict__ w0,
                           const float* __restrict__ b0, const float* __restrict__ bng,
                           const float* __restrict__ bnb, float* __restrict__ h0) {
  int idx = blockIdx.x * blockDim.x + threadIdx.x;     // b*128*4096
  if (idx >= BATCH * DM * L0) return;
  int lo = idx & (L0 - 1);
  int c  = (idx >> 12) & (DM - 1);
  int b  = idx >> 19;
  float acc = b0[c];
  int start = 2 * lo - 4;
  #pragma unroll
  for (int ci = 0; ci < 4; ++ci) {
    const float* xr = x + ((long)(b * 4 + ci)) * L_IN;
    const float* wr = w0 + (c * 4 + ci) * 9;
    #pragma unroll
    for (int t = 0; t < 9; ++t) {
      int p = start + t;
      if ((unsigned)p < (unsigned)L_IN) acc += xr[p] * wr[t];
    }
  }
  acc = fmaxf(acc, 0.0f);
  h0[idx] = acc * (bng[c] * rsqrtf(1.0f + 1e-5f)) + bnb[c];
}

__global__ __launch_bounds__(256) void stem_conv1(
    const float* __restrict__ h0, const float* __restrict__ w1, const float* __restrict__ b1,
    const float* __restrict__ bng, const float* __restrict__ bnb, float* __restrict__ h) {
  int b = blockIdx.x >> 7;
  int d = blockIdx.x & (DM - 1);
  __shared__ float wl[DM * 9];                         // 1152 floats
  for (int i = threadIdx.x; i < DM * 9; i += 256) wl[i] = w1[d * DM * 9 + i];
  __syncthreads();
  float bias = b1[d];
  float s  = bng[d] * rsqrtf(1.0f + 1e-5f);
  float bb = bnb[d];
  const float* xb = h0 + (long)b * DM * L0;
  for (int li = threadIdx.x; li < LH; li += 256) {
    int start = 2 * li - 4;
    float acc = bias;
    for (int ci = 0; ci < DM; ++ci) {
      const float* xr = xb + (long)ci * L0;
      const float* wr = wl + ci * 9;
      #pragma unroll
      for (int t = 0; t < 9; ++t) {
        int p = start + t;
        if ((unsigned)p < (unsigned)L0) acc += xr[p] * wr[t];
      }
    }
    acc = fmaxf(acc, 0.0f);
    h[((long)b * LH + li) * DM + d] = acc * s + bb;
  }
}

// ---------------- layer: LN + in_proj (WMMA bf16, TDM weight staging) -------
// Grid (ROWS/64, 512/64), block 256 (8 wave32s). 64x64 tile per block, K=128.
__global__ __launch_bounds__(256) void ln_inproj_wmma(
    const float* __restrict__ h, const unsigned short* __restrict__ Wb,
    const float* __restrict__ lng, const float* __restrict__ lnb,
    float* __restrict__ xz) {
  __shared__ unsigned short As[64][136];   // +8 pad: stride 272B -> conflict-free
  __shared__ unsigned short Bs[64][136];
  const int tid   = threadIdx.x;
  const int mBase = blockIdx.x * 64;
  const int nBase = blockIdx.y * 64;

  if (tid < 32) {                           // wave 0: TDM DMA of 64x128 bf16 weight tile
    tdm_load_tile_bf16(Wb + (size_t)nBase * DM, &Bs[0][0],
                       /*tile_d0=*/DM, /*tile_d1=*/64, /*stride=*/DM,
                       /*pad_interval(64DW)=*/5, /*pad_amount(4DW)=*/3, /*pad_en=*/1);
  }
  if (tid < 64) {                           // LayerNorm one row per thread
    const float* row = h + (long)(mBase + tid) * DM;
    float s = 0.f;
    for (int k = 0; k < DM; ++k) s += row[k];
    float mu = s * (1.0f / DM);
    float v = 0.f;
    for (int k = 0; k < DM; ++k) { float dd = row[k] - mu; v += dd * dd; }
    float rstd = rsqrtf(v * (1.0f / DM) + 1e-5f);
    for (int k = 0; k < DM; ++k)
      As[tid][k] = bf16bits((row[k] - mu) * rstd * lng[k] + lnb[k]);
  }
  if (tid < 32) __builtin_amdgcn_s_wait_tensorcnt(0);   // issuing wave drains TDM
  __syncthreads();                                      // releases all waves

  const int wave = tid >> 5, lane = tid & 31;
  const int mSub = wave & 3;
  const int half = lane >> 4, ln15 = lane & 15;
  for (int np = 0; np < 2; ++np) {
    int nSub = ((wave >> 2) << 1) + np;
    v8f c = {};
    #pragma unroll
    for (int k0 = 0; k0 < DM; k0 += 32) {
      v16bf a, bf;
      #pragma unroll
      for (int j = 0; j < 16; ++j) {
        int ka = k0 + j + ((j >= 8) ? 8 : 0) + half * 8;   // A 16x32 bf16 layout
        a[j]  = __builtin_bit_cast(__bf16, As[mSub * 16 + ln15][ka]);
        int kb = k0 + j + ((j >= 8) ? 8 : 0) + half * 8;   // B 32x16 bf16 layout
        bf[j] = __builtin_bit_cast(__bf16, Bs[nSub * 16 + ln15][kb]);
      }
      c = __builtin_amdgcn_wmma_f32_16x16x32_bf16(false, a, false, bf, (short)0, c, false, false);
    }
    #pragma unroll
    for (int r = 0; r < 8; ++r) {                            // D layout: M=r+8*half
      int row = mBase + mSub * 16 + r + half * 8;
      int col = nBase + nSub * 16 + ln15;
      xz[(long)row * 512 + col] = c[r];
    }
  }
}

// ---------------- layer: depthwise causal conv + SiLU ----------------
__global__ void dwconv_silu(const float* __restrict__ xz, const float* __restrict__ cw,
                            const float* __restrict__ cb, float* __restrict__ xc) {
  int idx = blockIdx.x * blockDim.x + threadIdx.x;   // ROWS*DI
  if (idx >= ROWS * DI) return;
  int d = idx & (DI - 1);
  int l = (idx >> 8) & (LH - 1);
  int b = idx >> 19;
  float acc = cb[d];
  #pragma unroll
  for (int t = 0; t < 4; ++t) {
    int ll = l - 3 + t;
    if (ll >= 0) acc += xz[((long)(b * LH + ll)) * 512 + d] * cw[d * 4 + t];
  }
  xc[idx] = siluf_(acc);
}

// ---------------- layer: x_proj (16384 x 40, K=256) ----------------
__global__ void xproj_k(const float* __restrict__ xc, const float* __restrict__ xw,
                        float* __restrict__ xdbl) {
  int idx = blockIdx.x * blockDim.x + threadIdx.x;
  if (idx >= ROWS * 40) return;
  int n = idx % 40;
  long row = idx / 40;
  const float* xr = xc + row * DI;
  const float* wr = xw + n * DI;
  float acc = 0.f;
  for (int k = 0; k < DI; ++k) acc += xr[k] * wr[k];
  xdbl[idx] = acc;
}

// ---------------- layer: dt_proj + softplus ----------------
__global__ void dtproj_k(const float* __restrict__ xdbl, const float* __restrict__ dtw,
                         const float* __restrict__ dtb, float* __restrict__ delta) {
  int idx = blockIdx.x * blockDim.x + threadIdx.x;   // ROWS*DI
  if (idx >= ROWS * DI) return;
  int d = idx & (DI - 1);
  long row = idx >> 8;
  float acc = dtb[d];
  #pragma unroll
  for (int j = 0; j < DTR; ++j) acc += xdbl[row * 40 + j] * dtw[d * DTR + j];
  delta[idx] = softplusf_(acc);
}

// ---------------- layer: selective scan ----------------
// Block per batch, thread per inner channel; 16 states in registers.
__global__ __launch_bounds__(256) void scan_k(
    const float* __restrict__ xc, const float* __restrict__ delta,
    const float* __restrict__ xdbl, const float* __restrict__ Alog,
    const float* __restrict__ Dp, float* __restrict__ y) {
  int b = blockIdx.x, d = threadIdx.x;
  float Av[DS];
  #pragma unroll
  for (int n = 0; n < DS; ++n) Av[n] = -__expf(Alog[d * DS + n]);
  float hs[DS] = {0.f};
  float Dv = Dp[d];
  __shared__ float BC[64][32];                 // B (16) + C (16) per timestep
  const long base = (long)b * LH;
  for (int l0 = 0; l0 < LH; l0 += 64) {
    for (int idx = threadIdx.x; idx < 64 * 32; idx += 256) {
      int ll = idx >> 5, n = idx & 31;
      BC[ll][n] = xdbl[(base + l0 + ll) * 40 + DTR + n];
    }
    __syncthreads();
    if (l0 + 64 < LH) __builtin_prefetch(&xdbl[(base + l0 + 64) * 40], 0, 1);
    for (int lc = 0; lc < 64; ++lc) {
      long ro = (base + l0 + lc) * DI + d;
      float dl = delta[ro];
      float u  = xc[ro];
      float du = dl * u;
      float acc = 0.f;
      #pragma unroll
      for (int n = 0; n < DS; ++n) {
        float dA = __expf(dl * Av[n]);
        hs[n] = dA * hs[n] + du * BC[lc][n];
        acc += hs[n] * BC[lc][DS + n];
      }
      y[ro] = acc + u * Dv;
    }
    __syncthreads();
  }
}

// ---------------- layer: gate * out_proj + residual (WMMA bf16 + TDM) -------
// Grid (ROWS/64, 128/64), block 256. K=256 processed in two 128-wide LDS chunks.
__global__ __launch_bounds__(256) void gate_outproj_wmma(
    const float* __restrict__ y, const float* __restrict__ xz,
    const unsigned short* __restrict__ Wb, float* __restrict__ h) {
  __shared__ unsigned short As[64][136];
  __shared__ unsigned short Bs[64][136];
  const int tid   = threadIdx.x;
  const int mBase = blockIdx.x * 64;
  const int nBase = blockIdx.y * 64;
  const int wave = tid >> 5, lane = tid & 31;
  const int mSub = wave & 3;
  const int half = lane >> 4, ln15 = lane & 15;
  v8f acc0 = {}, acc1 = {};

  for (int kc = 0; kc < DI; kc += 128) {
    if (tid < 32) {                      // TDM: 64x128 bf16 strided weight sub-tile
      tdm_load_tile_bf16(Wb + (size_t)nBase * DI + kc, &Bs[0][0],
                         /*tile_d0=*/128, /*tile_d1=*/64, /*stride=*/DI,
                         /*pad_interval(64DW)=*/5, /*pad_amount(4DW)=*/3, /*pad_en=*/1);
    }
    for (int idx = tid; idx < 64 * 128; idx += 256) {   // gated activations -> bf16
      int r = idx >> 7, k = idx & 127;
      long row = mBase + r;
      float zv = xz[row * 512 + DI + kc + k];
      As[r][k] = bf16bits(y[row * DI + kc + k] * siluf_(zv));
    }
    if (tid < 32) __builtin_amdgcn_s_wait_tensorcnt(0);
    __syncthreads();
    for (int np = 0; np < 2; ++np) {
      int nSub = ((wave >> 2) << 1) + np;
      v8f c = (np == 0) ? acc0 : acc1;
      #pragma unroll
      for (int k0 = 0; k0 < 128; k0 += 32) {
        v16bf a, bf;
        #pragma unroll
        for (int j = 0; j < 16; ++j) {
          int ka = k0 + j + ((j >= 8) ? 8 : 0) + half * 8;
          a[j]  = __builtin_bit_cast(__bf16, As[mSub * 16 + ln15][ka]);
          int kb = k0 + j + ((j >= 8) ? 8 : 0) + half * 8;
          bf[j] = __builtin_bit_cast(__bf16, Bs[nSub * 16 + ln15][kb]);
        }
        c = __builtin_amdgcn_wmma_f32_16x16x32_bf16(false, a, false, bf, (short)0, c, false, false);
      }
      if (np == 0) acc0 = c; else acc1 = c;
    }
    __syncthreads();
  }
  for (int np = 0; np < 2; ++np) {
    int nSub = ((wave >> 2) << 1) + np;
    v8f c = (np == 0) ? acc0 : acc1;
    #pragma unroll
    for (int r = 0; r < 8; ++r) {
      int row = mBase + mSub * 16 + r + half * 8;
      int col = nBase + nSub * 16 + ln15;
      h[(long)row * DM + col] += c[r];          // residual add
    }
  }
}

// ---------------- memory path ----------------
__global__ void seq_mean_k(const float* __restrict__ h, float* __restrict__ seq) {
  int b = blockIdx.x, d = threadIdx.x;          // block 128
  float s = 0.f;
  for (int l = 0; l < LH; ++l) s += h[((long)b * LH + l) * DM + d];
  seq[b * DM + d] = s * (1.0f / LH);
}

__global__ void mem_update_k(const float* __restrict__ seq,
                             const float* __restrict__ gw, const float* __restrict__ gb,
                             const float* __restrict__ uw, const float* __restrict__ ub,
                             float* __restrict__ mem) {
  int b = blockIdx.x, d = threadIdx.x;          // block 128
  __shared__ float comb[2 * DM];
  comb[d]      = seq[b * DM + d];
  comb[DM + d] = mem[b * DM + d];
  __syncthreads();
  float ga = gb[d], ua = ub[d];
  for (int k = 0; k < 2 * DM; ++k) {
    float cv = comb[k];
    ga += cv * gw[d * 2 * DM + k];
    ua += cv * uw[d * 2 * DM + k];
  }
  float g = sigmoidf_(ga);
  float u = tanhf(ua);
  float mo = comb[DM + d];
  mem[b * DM + d] = mo * (1.0f - g) + u * g;
}

__global__ void add_mem_k(float* __restrict__ h, const float* __restrict__ mem) {
  int idx = blockIdx.x * blockDim.x + threadIdx.x;   // ROWS*DM
  if (idx >= ROWS * DM) return;
  int d = idx & (DM - 1);
  int b = idx >> 18;                                  // LH*DM = 2^18
  h[idx] += mem[b * DM + d];
}

// ---------------- final LN + pool + classifier ----------------
__global__ void ln_final_k(const float* __restrict__ h, const float* __restrict__ g,
                           const float* __restrict__ bb, float* __restrict__ hf) {
  int row = blockIdx.x * blockDim.x + threadIdx.x;
  if (row >= ROWS) return;
  const float* r = h + (long)row * DM;
  float s = 0.f;
  for (int k = 0; k < DM; ++k) s += r[k];
  float mu = s * (1.0f / DM);
  float v = 0.f;
  for (int k = 0; k < DM; ++k) { float dd = r[k] - mu; v += dd * dd; }
  float rstd = rsqrtf(v * (1.0f / DM) + 1e-5f);
  for (int k = 0; k < DM; ++k) hf[(long)row * DM + k] = (r[k] - mu) * rstd * g[k] + bb[k];
}

__global__ void pool_k(const float* __restrict__ hf, float* __restrict__ pool) {
  int b = blockIdx.x, d = threadIdx.x;          // block 128
  float s = 0.f;
  for (int l = 0; l < LH; ++l) s += hf[((long)b * LH + l) * DM + d];
  pool[b * DM + d] = s * (1.0f / LH);
}

__global__ void cls_k(const float* __restrict__ pool, const float* __restrict__ cw,
                      const float* __restrict__ cb, float* __restrict__ out) {
  int idx = blockIdx.x * blockDim.x + threadIdx.x;
  if (idx >= BATCH * NC) return;
  int c = idx % NC, b = idx / NC;
  float acc = cb[c];
  for (int k = 0; k < DM; ++k) acc += pool[b * DM + k] * cw[c * DM + k];
  out[b * NC + c] = acc;
}

// ---------------- host launch ----------------
extern "C" void kernel_launch(void* const* d_in, const int* in_sizes, int n_in,
                              void* d_out, int out_size, void* d_ws, size_t ws_size,
                              hipStream_t stream) {
  const float* x        = (const float*)d_in[0];
  const float* tok_w0   = (const float*)d_in[1];
  const float* tok_b0   = (const float*)d_in[2];
  const float* bn_g0    = (const float*)d_in[3];
  const float* bn_b0    = (const float*)d_in[4];
  const float* tok_w1   = (const float*)d_in[5];
  const float* tok_b1   = (const float*)d_in[6];
  const float* bn_g1    = (const float*)d_in[7];
  const float* bn_b1    = (const float*)d_in[8];
  const float* ln_g     = (const float*)d_in[9];
  const float* ln_b     = (const float*)d_in[10];
  const float* in_proj  = (const float*)d_in[11];
  const float* conv_w   = (const float*)d_in[12];
  const float* conv_b   = (const float*)d_in[13];
  const float* xproj_w  = (const float*)d_in[14];
  const float* dtproj_w = (const float*)d_in[15];
  const float* dtproj_b = (const float*)d_in[16];
  const float* A_log    = (const float*)d_in[17];
  const float* Dp       = (const float*)d_in[18];
  const float* outproj  = (const float*)d_in[19];
  const float* gate_w   = (const float*)d_in[20];
  const float* gate_b   = (const float*)d_in[21];
  const float* upd_w    = (const float*)d_in[22];
  const float* upd_b    = (const float*)d_in[23];
  const float* norm_g   = (const float*)d_in[24];
  const float* norm_b   = (const float*)d_in[25];
  const float* cls_w    = (const float*)d_in[26];
  const float* cls_b    = (const float*)d_in[27];
  const float* mem_init = (const float*)d_in[28];
  float* out = (float*)d_out;

  // ---- workspace layout (~97 MB) ----
  char* w = (char*)d_ws;
  size_t off = 0;
  auto alloc = [&](size_t bytes) -> void* {
    void* p = (void*)(w + off);
    off += (bytes + 255) & ~(size_t)255;
    return p;
  };
  float* hbuf   = (float*)alloc((size_t)ROWS * DM * 4);   // persistent h (b,l,d)
  float* xzbuf  = (float*)alloc((size_t)ROWS * 512 * 4);  // xz; aliased as conv0 out in stem
  float* xcbuf  = (float*)alloc((size_t)ROWS * DI * 4);
  float* xdbl   = (float*)alloc((size_t)ROWS * 40 * 4);
  float* deltab = (float*)alloc((size_t)ROWS * DI * 4);   // reused as final LN buffer
  float* ybuf   = (float*)alloc((size_t)ROWS * DI * 4);
  float* seqb   = (float*)alloc((size_t)BATCH * DM * 4);
  float* memb   = (float*)alloc((size_t)BATCH * DM * 4);
  float* poolb  = (float*)alloc((size_t)BATCH * DM * 4);
  unsigned short* Winb  = (unsigned short*)alloc((size_t)NL * 512 * DM * 2);
  unsigned short* Woutb = (unsigned short*)alloc((size_t)NL * DM * DI * 2);
  float* h0 = xzbuf;   // stem conv0 output aliases xz region (stem runs first)

  // weights -> bf16 (cheap; redone every call for determinism)
  {
    int n1 = NL * 512 * DM, n2 = NL * DM * DI;
    cvt_bf16<<<(n1 + 255) / 256, 256, 0, stream>>>(in_proj, Winb, n1);
    cvt_bf16<<<(n2 + 255) / 256, 256, 0, stream>>>(outproj, Woutb, n2);
  }
  mem_init_k<<<(BATCH * DM + 255) / 256, 256, 0, stream>>>(mem_init, memb);

  // stem
  stem_conv0<<<(BATCH * DM * L0) / 256, 256, 0, stream>>>(x, tok_w0, tok_b0, bn_g0, bn_b0, h0);
  stem_conv1<<<BATCH * DM, 256, 0, stream>>>(h0, tok_w1, tok_b1, bn_g1, bn_b1, hbuf);

  // mamba layers
  for (int li = 0; li < NL; ++li) {
    ln_inproj_wmma<<<dim3(ROWS / 64, 512 / 64), 256, 0, stream>>>(
        hbuf, Winb + (size_t)li * 512 * DM, ln_g + li * DM, ln_b + li * DM, xzbuf);
    dwconv_silu<<<(ROWS * DI) / 256, 256, 0, stream>>>(
        xzbuf, conv_w + (size_t)li * DI * 4, conv_b + li * DI, xcbuf);
    xproj_k<<<(ROWS * 40 + 255) / 256, 256, 0, stream>>>(
        xcbuf, xproj_w + (size_t)li * 40 * DI, xdbl);
    dtproj_k<<<(ROWS * DI) / 256, 256, 0, stream>>>(
        xdbl, dtproj_w + (size_t)li * DI * DTR, dtproj_b + li * DI, deltab);
    scan_k<<<BATCH, DI, 0, stream>>>(
        xcbuf, deltab, xdbl, A_log + (size_t)li * DI * DS, Dp + li * DI, ybuf);
    gate_outproj_wmma<<<dim3(ROWS / 64, DM / 64), 256, 0, stream>>>(
        ybuf, xzbuf, Woutb + (size_t)li * DM * DI, hbuf);
    seq_mean_k<<<BATCH, DM, 0, stream>>>(hbuf, seqb);
    mem_update_k<<<BATCH, DM, 0, stream>>>(
        seqb, gate_w + (size_t)li * DM * 2 * DM, gate_b + li * DM,
        upd_w + (size_t)li * DM * 2 * DM, upd_b + li * DM, memb);
    add_mem_k<<<(ROWS * DM) / 256, 256, 0, stream>>>(hbuf, memb);
  }

  // final LN + mean pool + classifier
  ln_final_k<<<(ROWS + 255) / 256, 256, 0, stream>>>(hbuf, norm_g, norm_b, deltab);
  pool_k<<<BATCH, DM, 0, stream>>>(deltab, poolb);
  cls_k<<<(BATCH * NC + 255) / 256, 256, 0, stream>>>(poolb, cls_w, cls_b, out);
}
